// SpatialNeighborhoodAttention_88725434401380
// MI455X (gfx1250) — matrix operand order repaired and checked by reading, not compiled, not run on hardware
//
#include <hip/hip_runtime.h>
#include <hip/hip_bf16.h>

typedef __attribute__((ext_vector_type(16))) _Float16 v16h;
typedef __attribute__((ext_vector_type(2)))  __fp16   v2h;   // cvt_pkrtz result type
typedef __attribute__((ext_vector_type(8)))  float    v8f;

// Sizes: B=2, C=256, T=8, H=W=32 -> BT=16, N=1024, QKV=768, heads=8, hd=32

// ---------------------------------------------------------------------------
// Kernel 1: qkv[bt][n][0:768] = x_r[bt][n][0:256] @ Win^T + bin   (f16 out)
// x layout: x[((b*256 + c)*8 + t)*1024 + n]   (+1 channel => +8192 floats)
// Tiling: 64(M) x 64(J), K staged in chunks of 128 through LDS.
// Staging uses packed f32->f16 converts (v_cvt_pk_rtz_f16_f32) + b64 LDS stores.
// ---------------------------------------------------------------------------
__global__ __launch_bounds__(256) void qkv_gemm_f16wmma(
    const float* __restrict__ x, const float* __restrict__ w,
    const float* __restrict__ bias, _Float16* __restrict__ qkv) {
  __shared__ _Float16 Alds[64 * 128];
  __shared__ _Float16 Blds[64 * 128];

  const int bx = blockIdx.x;
  const int jt = bx % 12;            // 768 / 64
  const int mt = (bx / 12) % 16;     // 1024 / 64
  const int bt = bx / 192;           // 16 planes
  const int b = bt >> 3, t = bt & 7;
  const int n0 = mt * 64, j0 = jt * 64;

  const int tid   = threadIdx.x;
  const int lane  = tid & 31;
  const int wave  = tid >> 5;
  const int mi    = wave & 3;            // M subtile 0..3
  const int njb   = (wave >> 2) * 2;     // N subtile pair base
  const int lrow  = lane & 15;
  const int khalf = lane >> 4;

  // staging maps
  const int a_nl = tid & 63;             // spatial row (coalesced over n)
  const int a_g  = (tid >> 6) * 4;       // channel sub-offset {0,4,8,12}
  const int b_c4 = (tid & 31) * 4;       // contiguous 4-channel group
  const int b_j  = tid >> 5;             // weight row

  v8f acc0 = {0.f,0.f,0.f,0.f,0.f,0.f,0.f,0.f};
  v8f acc1 = {0.f,0.f,0.f,0.f,0.f,0.f,0.f,0.f};

  for (int kc = 0; kc < 256; kc += 128) {
    // Stage A: 64 n-rows x 128 channels, packed converts
    #pragma unroll
    for (int s = 0; s < 8; ++s) {
      int cl = a_g + s * 16;                     // covers all multiples of 4 in 0..124
      int c  = kc + cl;
      const float* xp = &x[((size_t)(b * 256 + c) * 8 + t) * 1024 + n0 + a_nl];
      v2h p0 = __builtin_amdgcn_cvt_pkrtz(xp[0],     xp[8192]);
      v2h p1 = __builtin_amdgcn_cvt_pkrtz(xp[16384], xp[24576]);
      *(v2h*)&Alds[a_nl * 128 + cl]     = p0;
      *(v2h*)&Alds[a_nl * 128 + cl + 2] = p1;
    }
    // Stage B: 64 j-rows x 128 channels, float4 loads + packed converts
    #pragma unroll
    for (int s = 0; s < 8; ++s) {
      int jj = b_j + s * 8;
      const float4 f = *(const float4*)&w[(size_t)(j0 + jj) * 256 + kc + b_c4];
      v2h p0 = __builtin_amdgcn_cvt_pkrtz(f.x, f.y);
      v2h p1 = __builtin_amdgcn_cvt_pkrtz(f.z, f.w);
      *(v2h*)&Blds[jj * 128 + b_c4]     = p0;
      *(v2h*)&Blds[jj * 128 + b_c4 + 2] = p1;
    }
    __syncthreads();

    #pragma unroll
    for (int k0 = 0; k0 < 128; k0 += 32) {
      v16h a, b0, b1;
      const int m = mi * 16 + lrow;
      #pragma unroll
      for (int v = 0; v < 8; ++v) {
        int kb = k0 + ((v >> 2) << 4) + (khalf << 3) + ((v & 3) << 1);
        a[2 * v]     = Alds[m * 128 + kb];
        a[2 * v + 1] = Alds[m * 128 + kb + 1];
      }
      const int col0 = njb * 16 + lrow;
      const int col1 = col0 + 16;
      #pragma unroll
      for (int v = 0; v < 8; ++v) {
        int kk = k0 + ((v >> 2) << 4) + (khalf << 3) + ((v & 3) << 1);
        b0[2 * v]     = Blds[col0 * 128 + kk];
        b0[2 * v + 1] = Blds[col0 * 128 + kk + 1];
        b1[2 * v]     = Blds[col1 * 128 + kk];
        b1[2 * v + 1] = Blds[col1 * 128 + kk + 1];
      }
      acc0 = __builtin_amdgcn_wmma_f32_16x16x32_f16(false, a, false, b0,
                                                    (short)0, acc0, false, false);
      acc1 = __builtin_amdgcn_wmma_f32_16x16x32_f16(false, a, false, b1,
                                                    (short)0, acc1, false, false);
    }
    __syncthreads();
  }

  const int jg0 = j0 + njb * 16 + lrow;
  const int jg1 = jg0 + 16;
  const float bj0 = bias[jg0], bj1 = bias[jg1];
  const size_t base = (size_t)bt * 1024 * 768;
  #pragma unroll
  for (int v = 0; v < 8; ++v) {
    int row = n0 + mi * 16 + khalf * 8 + v;
    qkv[base + (size_t)row * 768 + jg0] = (_Float16)(acc0[v] + bj0);
    qkv[base + (size_t)row * 768 + jg1] = (_Float16)(acc1[v] + bj1);
  }
}

// ---------------------------------------------------------------------------
// Kernel 2: 4-neighbor masked attention. 1 thread per (bt, head, query).
// qkv[bt][n][768]: q @ +0, k @ +256, v @ +512 (per-head stride 32).
// ---------------------------------------------------------------------------
__global__ __launch_bounds__(256) void neighborhood_attn(
    const _Float16* __restrict__ qkv, _Float16* __restrict__ out) {
  const int g    = blockIdx.x * 256 + threadIdx.x;  // 131072 total
  const int n    = g & 1023;
  const int head = (g >> 10) & 7;
  const int bt   = g >> 13;
  const size_t plane = (size_t)bt * 1024 * 768;

  const _Float16* qp = qkv + plane + (size_t)n * 768 + head * 32;
  float qr[32];
  #pragma unroll
  for (int d = 0; d < 32; ++d) qr[d] = (float)qp[d];

  const int r = n >> 5, cc = n & 31;
  int nbr[4]; int cnt = 0;
  if (r  > 0)  nbr[cnt++] = n - 32;
  if (r  < 31) nbr[cnt++] = n + 32;
  if (cc > 0)  nbr[cnt++] = n - 1;
  if (cc < 31) nbr[cnt++] = n + 1;

  float sc[4];
  float mx = -3.4e38f;
  for (int i = 0; i < cnt; ++i) {
    const _Float16* kp = qkv + plane + (size_t)nbr[i] * 768 + 256 + head * 32;
    float dot = 0.f;
    #pragma unroll
    for (int d = 0; d < 32; ++d) dot += qr[d] * (float)kp[d];
    sc[i] = dot * 0.17677669529663687f;  // 1/sqrt(32)
    mx = fmaxf(mx, sc[i]);
  }
  float sum = 0.f;
  for (int i = 0; i < cnt; ++i) { sc[i] = __expf(sc[i] - mx); sum += sc[i]; }
  const float inv = 1.f / sum;

  float o[32];
  #pragma unroll
  for (int d = 0; d < 32; ++d) o[d] = 0.f;
  for (int i = 0; i < cnt; ++i) {
    const _Float16* vp = qkv + plane + (size_t)nbr[i] * 768 + 512 + head * 32;
    const float p = sc[i] * inv;
    #pragma unroll
    for (int d = 0; d < 32; ++d) o[d] += p * (float)vp[d];
  }
  _Float16* op = out + (size_t)bt * 1024 * 256 + (size_t)n * 256 + head * 32;
  #pragma unroll
  for (int d = 0; d < 32; ++d) op[d] = (_Float16)o[d];
}

// ---------------------------------------------------------------------------
// Kernel 3: out-proj GEMM + scatter to (B, C, T, H, W) f32 output.
// A-tile is already f16 in the workspace -> stage it with the CDNA5 async
// DMA path (global_load_async_to_lds_b128, ASYNCcnt), overlapping with the
// VGPR-path f32->f16 convert of the weight tile.
// ---------------------------------------------------------------------------
__global__ __launch_bounds__(256) void outproj_gemm_f16wmma(
    const _Float16* __restrict__ attn, const float* __restrict__ w,
    const float* __restrict__ bias, float* __restrict__ out) {
  __shared__ _Float16 Alds[64 * 128];
  __shared__ _Float16 Blds[64 * 128];

  const int bx = blockIdx.x;
  const int jt = bx & 3;             // 256 / 64
  const int mt = (bx >> 2) & 15;     // 1024 / 64
  const int bt = bx >> 6;
  const int b = bt >> 3, t = bt & 7;
  const int n0 = mt * 64, j0 = jt * 64;

  const int tid   = threadIdx.x;
  const int lane  = tid & 31;
  const int wave  = tid >> 5;
  const int mi    = wave & 3;
  const int njb   = (wave >> 2) * 2;
  const int lrow  = lane & 15;
  const int khalf = lane >> 4;

  const int b_c4 = (tid & 31) * 4;
  const int b_j  = tid >> 5;

  const unsigned lds_a = (unsigned)(uintptr_t)&Alds[0];

  v8f acc0 = {0.f,0.f,0.f,0.f,0.f,0.f,0.f,0.f};
  v8f acc1 = {0.f,0.f,0.f,0.f,0.f,0.f,0.f,0.f};

  for (int kc = 0; kc < 256; kc += 128) {
    // --- Async-DMA stage of A (f16, no conversion): 64 rows x 256B each.
    // Each lane moves 16B; 1024 chunks over 256 threads = 4 instructions.
    const char* gbase =
        (const char*)(attn + (size_t)bt * 1024 * 256 + (size_t)n0 * 256 + kc);
    #pragma unroll
    for (int i = 0; i < 4; ++i) {
      int idx = i * 256 + tid;          // 0..1023
      int nl  = idx >> 4;               // row 0..63
      int c16 = idx & 15;               // 16B chunk within 256B row slice
      unsigned ldsaddr = lds_a + (unsigned)(nl * 256 + c16 * 16);
      const char* gp = gbase + (size_t)nl * 512 + c16 * 16;
      asm volatile("global_load_async_to_lds_b128 %0, %1, off"
                   :: "v"(ldsaddr), "v"(gp)
                   : "memory");
    }

    // --- VGPR-path stage of B (f32 weights -> f16), overlaps the DMA.
    #pragma unroll
    for (int s = 0; s < 8; ++s) {
      int jj = b_j + s * 8;
      const float4 f = *(const float4*)&w[(size_t)(j0 + jj) * 256 + kc + b_c4];
      v2h p0 = __builtin_amdgcn_cvt_pkrtz(f.x, f.y);
      v2h p1 = __builtin_amdgcn_cvt_pkrtz(f.z, f.w);
      *(v2h*)&Blds[jj * 128 + b_c4]     = p0;
      *(v2h*)&Blds[jj * 128 + b_c4 + 2] = p1;
    }

    asm volatile("s_wait_asynccnt 0" ::: "memory");
    __syncthreads();

    #pragma unroll
    for (int k0 = 0; k0 < 128; k0 += 32) {
      v16h a, b0, b1;
      const int m = mi * 16 + lrow;
      #pragma unroll
      for (int v = 0; v < 8; ++v) {
        int kb = k0 + ((v >> 2) << 4) + (khalf << 3) + ((v & 3) << 1);
        a[2 * v]     = Alds[m * 128 + kb];
        a[2 * v + 1] = Alds[m * 128 + kb + 1];
      }
      const int col0 = njb * 16 + lrow;
      const int col1 = col0 + 16;
      #pragma unroll
      for (int v = 0; v < 8; ++v) {
        int kk = k0 + ((v >> 2) << 4) + (khalf << 3) + ((v & 3) << 1);
        b0[2 * v]     = Blds[col0 * 128 + kk];
        b0[2 * v + 1] = Blds[col0 * 128 + kk + 1];
        b1[2 * v]     = Blds[col1 * 128 + kk];
        b1[2 * v + 1] = Blds[col1 * 128 + kk + 1];
      }
      acc0 = __builtin_amdgcn_wmma_f32_16x16x32_f16(false, a, false, b0,
                                                    (short)0, acc0, false, false);
      acc1 = __builtin_amdgcn_wmma_f32_16x16x32_f16(false, a, false, b1,
                                                    (short)0, acc1, false, false);
    }
    __syncthreads();
  }

  const int jg0 = j0 + njb * 16 + lrow;
  const int jg1 = jg0 + 16;
  const float bj0 = bias[jg0], bj1 = bias[jg1];
  #pragma unroll
  for (int v = 0; v < 8; ++v) {
    int row = n0 + mi * 16 + khalf * 8 + v;  // spatial n
    out[((size_t)(b * 256 + jg0) * 8 + t) * 1024 + row] = acc0[v] + bj0;
    out[((size_t)(b * 256 + jg1) * 8 + t) * 1024 + row] = acc1[v] + bj1;
  }
}

extern "C" void kernel_launch(void* const* d_in, const int* in_sizes, int n_in,
                              void* d_out, int out_size, void* d_ws, size_t ws_size,
                              hipStream_t stream) {
  const float* x     = (const float*)d_in[0];  // (2,256,8,32,32)
  const float* w_in  = (const float*)d_in[1];  // (768,256)
  const float* b_in  = (const float*)d_in[2];  // (768,)
  const float* w_out = (const float*)d_in[3];  // (256,256)
  const float* b_out = (const float*)d_in[4];  // (256,)
  float* out = (float*)d_out;                  // (2,256,8,32,32)

  _Float16* qkv_ws  = (_Float16*)d_ws;                      // 16*1024*768 f16
  _Float16* attn_ws = qkv_ws + (size_t)16 * 1024 * 768;     // 16*1024*256 f16

  qkv_gemm_f16wmma<<<3072, 256, 0, stream>>>(x, w_in, b_in, qkv_ws);
  neighborhood_attn<<<512, 256, 0, stream>>>(qkv_ws, attn_ws);
  outproj_gemm_f16wmma<<<1024, 256, 0, stream>>>(attn_ws, w_out, b_out, out);
}
